// MultiHeadAttention_69226282877604
// MI455X (gfx1250) — compile-verified, matrix operand order checked by reading, not run on hardware
//
#include <hip/hip_runtime.h>

// ---------------------------------------------------------------------------
// MHA forward for MI455X (gfx1250, wave32, WMMA bf16 16x16x32, f32 accum)
// B=4, T=2048, D=1024, H=16, hd=64
// ---------------------------------------------------------------------------

#define B_DIM 4
#define T_DIM 2048
#define D_DIM 1024
#define NH    16
#define HD    64
#define MROWS (B_DIM * T_DIM)   // 8192
#define KT    64                // keys per flash KV tile

typedef __attribute__((ext_vector_type(16))) __bf16 v16bf;
typedef __attribute__((ext_vector_type(8)))  __bf16 v8bf;
typedef __attribute__((ext_vector_type(8)))  float  v8f;

__device__ __forceinline__ __bf16 f2bf(float f) { return (__bf16)f; }

// ---------------------------------------------------------------------------
// Fragment load: the 16x32 bf16 A/B layout (ISA 7.12.2) puts each lane's 16
// elements in two contiguous 16-byte chunks of its source row.
// ---------------------------------------------------------------------------
struct bf16x16_bits { uint4 lo, hi; };

__device__ __forceinline__ v16bf load_frag(const __bf16* row, int half) {
    bf16x16_bits t;
    t.lo = *(const uint4*)(row + half * 8);
    t.hi = *(const uint4*)(row + 16 + half * 8);
    return __builtin_bit_cast(v16bf, t);
}

// ---------------------------------------------------------------------------
// Max-reduce across each 16-lane half using DPP only (no LDS bpermute).
// ---------------------------------------------------------------------------
__device__ __forceinline__ float rowmax16(float v) {
    int x;
    x = __builtin_amdgcn_update_dpp(0, __builtin_bit_cast(int, v), 0xB1,  0xf, 0xf, true);
    v = fmaxf(v, __builtin_bit_cast(float, x));
    x = __builtin_amdgcn_update_dpp(0, __builtin_bit_cast(int, v), 0x4E,  0xf, 0xf, true);
    v = fmaxf(v, __builtin_bit_cast(float, x));
    x = __builtin_amdgcn_update_dpp(0, __builtin_bit_cast(int, v), 0x141, 0xf, 0xf, true);
    v = fmaxf(v, __builtin_bit_cast(float, x));
    x = __builtin_amdgcn_update_dpp(0, __builtin_bit_cast(int, v), 0x140, 0xf, 0xf, true);
    v = fmaxf(v, __builtin_bit_cast(float, x));
    return v;
}

// ---------------------------------------------------------------------------
// f32 -> bf16 conversion (vector-4)
// ---------------------------------------------------------------------------
__global__ __launch_bounds__(256) void cvt_f32_bf16(const float* __restrict__ src,
                                                    __bf16* __restrict__ dst, int n4) {
    int i = blockIdx.x * blockDim.x + threadIdx.x;
    if (i < n4) {
        float4 f = ((const float4*)src)[i];
        struct H4 { __bf16 a, b, c, d; } h{f2bf(f.x), f2bf(f.y), f2bf(f.z), f2bf(f.w)};
        *(H4*)(dst + 4 * (size_t)i) = h;
    }
}

// ---------------------------------------------------------------------------
// C[M,N] = A[M,K] @ W[N,K]^T   (bf16 in, f32 accum)
// ---------------------------------------------------------------------------
template <bool F32OUT>
__global__ __launch_bounds__(256) void gemm_bt(const __bf16* __restrict__ A,
                                               const __bf16* __restrict__ W,
                                               const float* __restrict__ bias,
                                               __bf16* __restrict__ Cbf,
                                               float* __restrict__ Cf,
                                               int M, int N, int K) {
    __shared__ __bf16 As[128][64 + 8];   // row stride 144B (16B aligned)
    __shared__ __bf16 Bs[128][64 + 8];

    const int tid  = threadIdx.x;
    const int lane = tid & 31, wave = tid >> 5;
    const int waveM = wave >> 2, waveN = wave & 3;
    const int half = lane >> 4, l16 = lane & 15;
    const int m0 = blockIdx.y * 128;
    const int n0 = blockIdx.x * 128;

    v8f acc[4][2] = {};

    for (int k0 = 0; k0 < K; k0 += 64) {
        {
            const int colg = tid & 7;      // 8 groups x 8 halfs = 64 cols
            const int row  = tid >> 3;     // 32 rows / pass
#pragma unroll
            for (int p = 0; p < 4; ++p) {
                int r = row + p * 32;
                *(uint4*)&As[r][colg * 8] =
                    *(const uint4*)&A[(size_t)(m0 + r) * K + k0 + colg * 8];
                *(uint4*)&Bs[r][colg * 8] =
                    *(const uint4*)&W[(size_t)(n0 + r) * K + k0 + colg * 8];
            }
        }
        __syncthreads();

#pragma unroll
        for (int kk = 0; kk < 2; ++kk) {
            v16bf a[4], b[2];
#pragma unroll
            for (int i = 0; i < 4; ++i)
                a[i] = load_frag(&As[waveM * 64 + i * 16 + l16][kk * 32], half);
#pragma unroll
            for (int j = 0; j < 2; ++j)
                b[j] = load_frag(&Bs[waveN * 32 + j * 16 + l16][kk * 32], half);

#pragma unroll
            for (int i = 0; i < 4; ++i)
#pragma unroll
                for (int j = 0; j < 2; ++j)
                    acc[i][j] = __builtin_amdgcn_wmma_f32_16x16x32_bf16(
                        false, a[i], false, b[j], (short)0, acc[i][j], false, false);
        }
        __syncthreads();
    }

#pragma unroll
    for (int i = 0; i < 4; ++i)
#pragma unroll
        for (int j = 0; j < 2; ++j)
#pragma unroll
            for (int r = 0; r < 8; ++r) {
                const int mi = m0 + waveM * 64 + i * 16 + r + 8 * half;
                const int ni = n0 + waveN * 32 + j * 16 + l16;
                const float v = acc[i][j][r];
                if (F32OUT)
                    Cf[(size_t)mi * N + ni] = v + bias[ni];
                else
                    Cbf[(size_t)mi * N + ni] = f2bf(v);
            }
}

// ---------------------------------------------------------------------------
// Flash attention (causal). Block = 128 threads (4 waves); each wave owns
// 32 q-rows (two 16-row fragment sets), block owns 128 q-rows. KV streamed
// in 64-key tiles through LDS. The two q-sets are processed SEQUENTIALLY
// inside a tile so each set's score/P fragments die before the next set
// starts (keeps allocation under 256 VGPRs; avoids s_set_vgpr_msb and
// pressure copies). Interior tiles run a mask-free path; only the single
// diagonal tile per wave runs the masked path (separate scalar branches).
// ---------------------------------------------------------------------------
__global__ __launch_bounds__(128) void flash_attn(const __bf16* __restrict__ Q,
                                                  const __bf16* __restrict__ Kq,
                                                  const __bf16* __restrict__ Vq,
                                                  __bf16* __restrict__ O) {
    __shared__ __bf16 Ks[KT][HD + 8];       // [key][d], row stride 144B
    __shared__ __bf16 Vt[HD][KT + 8];       // [d][key], row stride 144B
    __shared__ __bf16 Ps[4][16][KT];        // per-wave P re-layout (reused per set)

    const int tid  = threadIdx.x;
    const int lane = tid & 31;
    const int wave = __builtin_amdgcn_readfirstlane(tid >> 5);   // SGPR
    const int half = lane >> 4, l16 = lane & 15;
    const int bh = blockIdx.y;
    const int b  = bh / NH, h = bh % NH;
    const int qBlock = blockIdx.x * 128;
    const int qWave  = qBlock + wave * 32;  // SGPR; rows qWave..qWave+31
    const float sm_scale = 0.125f;          // 1/sqrt(64)

    const size_t rowBase = (size_t)b * T_DIM;
    const size_t hOff    = (size_t)h * HD;

    v8f  o[2][4] = {};
    float mrow[2][8], lrow[2][8];
#pragma unroll
    for (int s = 0; s < 2; ++s)
#pragma unroll
        for (int r = 0; r < 8; ++r) { mrow[s][r] = -1e30f; lrow[s][r] = 0.f; }

    v16bf ones;
#pragma unroll
    for (int e = 0; e < 16; ++e) ones[e] = (__bf16)1.0f;

    // Q fragments: two 16-row sets, each with 2 k-steps over hd
    v16bf qf[2][2];
#pragma unroll
    for (int set = 0; set < 2; ++set) {
        const __bf16* qp = &Q[(rowBase + qWave + set * 16 + l16) * D_DIM + hOff];
        qf[set][0] = load_frag(qp, half);
        qf[set][1] = load_frag(qp + 32, half);
    }

    // per-tile, per-set compute; `masked` is literal at each call site.
    auto set_body = [&](int k0, int set, bool masked) {
        // S = Q K^T : 4 key sub-tiles x 2 hd-steps
        v8f s[4] = {};
#pragma unroll
        for (int j = 0; j < 4; ++j)
#pragma unroll
            for (int st = 0; st < 2; ++st) {
                v16bf bfrag = load_frag(&Ks[j * 16 + l16][st * 32], half);
                s[j] = __builtin_amdgcn_wmma_f32_16x16x32_bf16(
                    false, qf[set][st], false, bfrag, (short)0, s[j], false, false);
            }

        const int qb = qWave + set * 16;
        float corr[8];
#pragma unroll
        for (int r = 0; r < 8; ++r) {
            const int q = qb + r + 8 * half;
            float sv[4];
#pragma unroll
            for (int j = 0; j < 4; ++j) {
                sv[j] = s[j][r] * sm_scale;
                if (masked) {
                    if (k0 + j * 16 + l16 > q) sv[j] = -1e30f;
                }
            }
            float mx = fmaxf(fmaxf(sv[0], sv[1]), fmaxf(sv[2], sv[3]));
            mx = rowmax16(mx);
            const float mnew = fmaxf(mrow[set][r], mx);
            corr[r] = __expf(mrow[set][r] - mnew);
            mrow[set][r] = mnew;
#pragma unroll
            for (int j = 0; j < 4; ++j) {
                const float p = __expf(sv[j] - mnew);
                Ps[wave][r + 8 * half][j * 16 + l16] = f2bf(p);
            }
#pragma unroll
            for (int nt = 0; nt < 4; ++nt) o[set][nt][r] *= corr[r];
        }

        // P fragments (A-layout over 64 keys -> 2 frags)
        v16bf pf0 = load_frag(&Ps[wave][l16][0], half);
        v16bf pf1 = load_frag(&Ps[wave][l16][32], half);

        // row-sums via ones-matrix WMMA chain
        v8f ls = {};
        ls = __builtin_amdgcn_wmma_f32_16x16x32_bf16(
            false, pf0, false, ones, (short)0, ls, false, false);
        ls = __builtin_amdgcn_wmma_f32_16x16x32_bf16(
            false, pf1, false, ones, (short)0, ls, false, false);
#pragma unroll
        for (int r = 0; r < 8; ++r)
            lrow[set][r] = lrow[set][r] * corr[r] + ls[r];

        // O += P V
#pragma unroll
        for (int nt = 0; nt < 4; ++nt) {
            v16bf v0 = load_frag(&Vt[nt * 16 + l16][0], half);
            o[set][nt] = __builtin_amdgcn_wmma_f32_16x16x32_bf16(
                false, pf0, false, v0, (short)0, o[set][nt], false, false);
            v16bf v1 = load_frag(&Vt[nt * 16 + l16][32], half);
            o[set][nt] = __builtin_amdgcn_wmma_f32_16x16x32_bf16(
                false, pf1, false, v1, (short)0, o[set][nt], false, false);
        }
    };

    const int kvEnd = qBlock + 128;         // block-uniform (barrier safety)
    for (int k0 = 0; k0 < kvEnd; k0 += KT) {
        {
            const int colg = tid & 7;    // 8 groups x 8 halfs = 64 cols
            const int row  = tid >> 3;   // 16 rows / pass
#pragma unroll
            for (int p = 0; p < 4; ++p) {
                int r = row + p * 16;
                *(uint4*)&Ks[r][colg * 8] =
                    *(const uint4*)&Kq[(rowBase + k0 + r) * D_DIM + hOff + colg * 8];
                uint4 vv = *(const uint4*)&Vq[(rowBase + k0 + r) * D_DIM + hOff + colg * 8];
                v8bf ve = __builtin_bit_cast(v8bf, vv);
#pragma unroll
                for (int e = 0; e < 8; ++e) Vt[colg * 8 + e][r] = ve[e];
            }
        }
        __syncthreads();

        if (k0 + KT - 1 <= qWave) {         // interior: mask-free path
            set_body(k0, 0, false);
            set_body(k0, 1, false);
        } else if (k0 <= qWave + 31) {      // single diagonal tile per wave
            set_body(k0, 0, true);
            set_body(k0, 1, true);
        }
        __syncthreads();
    }

    // finalize: O /= l, store bf16 to [B*T, D] at head offset
#pragma unroll
    for (int set = 0; set < 2; ++set)
#pragma unroll
        for (int r = 0; r < 8; ++r) {
            const float inv = 1.0f / lrow[set][r];
            const int q = qWave + set * 16 + r + 8 * half;
#pragma unroll
            for (int nt = 0; nt < 4; ++nt)
                O[(rowBase + q) * D_DIM + hOff + nt * 16 + l16] = f2bf(o[set][nt][r] * inv);
        }
}

// ---------------------------------------------------------------------------
// Host-side orchestration
// ---------------------------------------------------------------------------
extern "C" void kernel_launch(void* const* d_in, const int* in_sizes, int n_in,
                              void* d_out, int out_size, void* d_ws, size_t ws_size,
                              hipStream_t stream) {
    (void)in_sizes; (void)n_in; (void)out_size; (void)ws_size;
    const float* x  = (const float*)d_in[0];
    const float* Wq = (const float*)d_in[1];
    const float* Wk = (const float*)d_in[2];
    const float* Wv = (const float*)d_in[3];
    const float* Wo = (const float*)d_in[4];
    const float* bo = (const float*)d_in[5];
    float* out = (float*)d_out;

    const size_t nX = (size_t)MROWS * D_DIM;     // 8.4M elems
    const size_t nW = (size_t)D_DIM * D_DIM;     // 1.05M elems

    __bf16* xb  = (__bf16*)d_ws;                 // [8192,1024] (reused as attn-out)
    __bf16* wqb = xb + nX;
    __bf16* wkb = wqb + nW;
    __bf16* wvb = wkb + nW;
    __bf16* wob = wvb + nW;
    __bf16* Qb  = wob + nW;
    __bf16* Kb  = Qb + nX;
    __bf16* Vb  = Kb + nX;

    // 1) convert inputs to bf16
    {
        dim3 blk(256);
        cvt_f32_bf16<<<dim3((nX / 4 + 255) / 256), blk, 0, stream>>>(x, xb, (int)(nX / 4));
        cvt_f32_bf16<<<dim3((nW / 4 + 255) / 256), blk, 0, stream>>>(Wq, wqb, (int)(nW / 4));
        cvt_f32_bf16<<<dim3((nW / 4 + 255) / 256), blk, 0, stream>>>(Wk, wkb, (int)(nW / 4));
        cvt_f32_bf16<<<dim3((nW / 4 + 255) / 256), blk, 0, stream>>>(Wv, wvb, (int)(nW / 4));
        cvt_f32_bf16<<<dim3((nW / 4 + 255) / 256), blk, 0, stream>>>(Wo, wob, (int)(nW / 4));
    }

    // 2) Q/K/V projections: [8192,1024] = xb @ W^T
    {
        dim3 grid(D_DIM / 128, MROWS / 128), blk(256);
        gemm_bt<false><<<grid, blk, 0, stream>>>(xb, wqb, nullptr, Qb, nullptr,
                                                 MROWS, D_DIM, D_DIM);
        gemm_bt<false><<<grid, blk, 0, stream>>>(xb, wkb, nullptr, Kb, nullptr,
                                                 MROWS, D_DIM, D_DIM);
        gemm_bt<false><<<grid, blk, 0, stream>>>(xb, wvb, nullptr, Vb, nullptr,
                                                 MROWS, D_DIM, D_DIM);
    }

    // 3) flash attention -> attn output reuses xb (x no longer needed)
    {
        dim3 grid(T_DIM / 128, B_DIM * NH), blk(128);
        flash_attn<<<grid, blk, 0, stream>>>(Qb, Kb, Vb, xb);
    }

    // 4) output projection with bias -> f32 d_out
    {
        dim3 grid(D_DIM / 128, MROWS / 128), blk(256);
        gemm_bt<true><<<grid, blk, 0, stream>>>(xb, wob, bo, nullptr, out,
                                                MROWS, D_DIM, D_DIM);
    }
}
